// UltraStaticDecoderCached_84774064488710
// MI455X (gfx1250) — matrix operand order repaired and checked by reading, not compiled, not run on hardware
//
#include <hip/hip_runtime.h>
#include <hip/hip_bf16.h>

// ---------------- model constants (static shapes from reference) ----------------
#define LAYERS 8
#define HEADS  8
#define DH     128
#define DMODEL 1024
#define FFN    4096
#define VOCAB  16384
#define ENCLEN 438
#define ENCPAD 448      // 438 rounded up to 16
#define PASTLEN 447
#define TSELF   (PASTLEN + 1)   // 448
#define KCHUNK  128             // K-depth staged per LDS round in the WMMA GEMM
#define GEMV_KS 8               // split-K factor for GEMVs

typedef __attribute__((ext_vector_type(16))) _Float16 v16h;
typedef __attribute__((ext_vector_type(8)))  _Float16 v8h;
typedef __attribute__((ext_vector_type(8)))  float    v8f;
typedef __attribute__((ext_vector_type(4)))  int      v4i;

// ---------------- async global->LDS staging (gfx1250 ASYNCcnt path) ----------------
#if defined(__gfx1250__) && __has_builtin(__builtin_amdgcn_global_load_async_to_lds_b128) && __has_builtin(__builtin_amdgcn_s_wait_asynccnt)
#define HAVE_ASYNC_LDS 1
typedef __attribute__((address_space(1))) v4i gas_v4i;   // global int4
typedef __attribute__((address_space(3))) v4i las_v4i;   // LDS int4
#else
#define HAVE_ASYNC_LDS 0
#endif

// Copy nfloats (multiple of 4, 16B-aligned both sides) from global to LDS,
// then make it visible to the whole block. Uses GLOBAL_LOAD_ASYNC_TO_LDS_B128
// when available (tracked by ASYNCcnt), else plain loads.
__device__ __forceinline__ void stage_to_lds_f32(float* __restrict__ lds_dst,
                                                 const float* __restrict__ gsrc,
                                                 int nfloats) {
#if HAVE_ASYNC_LDS
    for (int i = threadIdx.x * 4; i < nfloats; i += blockDim.x * 4) {
        __builtin_amdgcn_global_load_async_to_lds_b128(
            (gas_v4i*)(gsrc + i),
            (las_v4i*)(lds_dst + i),
            0, 0);
    }
    __builtin_amdgcn_s_wait_asynccnt(0);
    __syncthreads();
#else
    for (int i = threadIdx.x; i < nfloats; i += blockDim.x)
        lds_dst[i] = gsrc[i];
    __syncthreads();
#endif
}

// ---------------- block reductions (wave32) ----------------
__device__ __forceinline__ float wave_sum(float v) {
    #pragma unroll
    for (int off = 16; off > 0; off >>= 1) v += __shfl_down(v, off, 32);
    return v;
}
__device__ __forceinline__ float wave_max(float v) {
    #pragma unroll
    for (int off = 16; off > 0; off >>= 1) v = fmaxf(v, __shfl_down(v, off, 32));
    return v;
}
__device__ float block_sum(float v, float* red) {
    const int lane = threadIdx.x & 31, wid = threadIdx.x >> 5;
    const int nw = (blockDim.x + 31) >> 5;
    v = wave_sum(v);
    __syncthreads();
    if (lane == 0) red[wid] = v;
    __syncthreads();
    float r = (threadIdx.x < (unsigned)nw) ? red[threadIdx.x] : 0.0f;
    if (wid == 0) r = wave_sum(r);
    if (threadIdx.x == 0) red[0] = r;
    __syncthreads();
    r = red[0];
    __syncthreads();
    return r;
}
__device__ float block_max(float v, float* red) {
    const int lane = threadIdx.x & 31, wid = threadIdx.x >> 5;
    const int nw = (blockDim.x + 31) >> 5;
    v = wave_max(v);
    __syncthreads();
    if (lane == 0) red[wid] = v;
    __syncthreads();
    float r = (threadIdx.x < (unsigned)nw) ? red[threadIdx.x] : -3.0e38f;
    if (wid == 0) r = wave_max(r);
    if (threadIdx.x == 0) red[0] = r;
    __syncthreads();
    r = red[0];
    __syncthreads();
    return r;
}

// ---------------- kernels ----------------

// x = emb[id] + pos[past_len]
__global__ void k_embed(const int* __restrict__ ids, const int* __restrict__ plen,
                        const float* __restrict__ emb, const float* __restrict__ pos,
                        float* __restrict__ x) {
    const int j = blockIdx.x * blockDim.x + threadIdx.x;
    if (j < DMODEL)
        x[j] = emb[(size_t)ids[0] * DMODEL + j] + pos[(size_t)plen[0] * DMODEL + j];
}

// out = LN(x) * g + b   (single block)
__global__ void k_layernorm(const float* __restrict__ x, const float* __restrict__ g,
                            const float* __restrict__ b, float* __restrict__ out, int D) {
    __shared__ float red[32];
    float s = 0.0f;
    for (int i = threadIdx.x; i < D; i += blockDim.x) s += x[i];
    const float mean = block_sum(s, red) / (float)D;
    float vv = 0.0f;
    for (int i = threadIdx.x; i < D; i += blockDim.x) { float d = x[i] - mean; vv += d * d; }
    const float var = block_sum(vv, red) / (float)D;
    const float rstd = rsqrtf(var + 1e-5f);
    for (int i = threadIdx.x; i < D; i += blockDim.x)
        out[i] = (x[i] - mean) * rstd * g[i] + b[i];
}

__device__ __forceinline__ float gelu_tanh(float x) {
    const float c = 0.7978845608028654f; // sqrt(2/pi)
    float t = tanhf(c * (x + 0.044715f * x * x * x));
    return 0.5f * x * (1.0f + t);
}

// split-K GEMV partials: part[ks*Nout + j] = sum_{i in chunk ks} h[i] * W[i*Nout+j]
// grid (Nout/256, GEMV_KS), block 256, dynamic LDS = (Din/GEMV_KS)*4 bytes
__global__ void k_gemv_part(const float* __restrict__ h, const float* __restrict__ W,
                            float* __restrict__ part, int Din, int Nout) {
    extern __shared__ __align__(16) float hs[];
    const int chunk = Din / GEMV_KS;
    const int i0 = blockIdx.y * chunk;
    stage_to_lds_f32(hs, h + i0, chunk);
    const int j = blockIdx.x * blockDim.x + threadIdx.x;
    if (j >= Nout) return;
    const float* __restrict__ Wp = W + (size_t)i0 * Nout + j;
    float acc = 0.0f;
    #pragma unroll 4
    for (int i = 0; i < chunk; ++i) {
        if ((i & 3) == 0)  // one cacheline-ahead hint per 4 rows -> global_prefetch_b8
            __builtin_prefetch(Wp + (size_t)(i + 4) * Nout, 0, 0);
        acc = fmaf(hs[i], Wp[(size_t)i * Nout], acc);
    }
    part[(size_t)blockIdx.y * Nout + j] = acc;
}

// deterministic fixed-order reduction + activation / residual
// mode: 0 = write, 1 = residual add, 2 = gelu then write
__global__ void k_gemv_fin(const float* __restrict__ part, float* __restrict__ out,
                           int Nout, int mode) {
    const int j = blockIdx.x * blockDim.x + threadIdx.x;
    if (j >= Nout) return;
    float acc = 0.0f;
    #pragma unroll
    for (int ks = 0; ks < GEMV_KS; ++ks) acc += part[(size_t)ks * Nout + j];
    if (mode == 1)      out[j] += acc;
    else if (mode == 2) out[j] = gelu_tanh(acc);
    else                out[j] = acc;
}

// self-attention for one head per block; o[h*DH+d]
__global__ void k_self_attn(const float* __restrict__ q, const float* __restrict__ knew,
                            const float* __restrict__ vnew,
                            const float* __restrict__ pastK, const float* __restrict__ pastV,
                            float* __restrict__ o) {
    const int h = blockIdx.x;
    __shared__ __align__(16) float qs[DH];
    __shared__ float sc[TSELF];
    __shared__ float red[32];
    stage_to_lds_f32(qs, q + h * DH, DH);
    const float scale = 0.08838834764831843f; // 1/sqrt(128)
    const float4* __restrict__ q4 = (const float4*)qs;
    for (int t = threadIdx.x; t < TSELF; t += blockDim.x) {
        const float* kp = (t < PASTLEN) ? (pastK + ((size_t)h * PASTLEN + t) * DH)
                                        : (knew + h * DH);
        const float4* __restrict__ k4 = (const float4*)kp;
        float s = 0.0f;
        #pragma unroll
        for (int d = 0; d < DH / 4; ++d) {
            float4 a = q4[d], b = k4[d];
            s = fmaf(a.x, b.x, s); s = fmaf(a.y, b.y, s);
            s = fmaf(a.z, b.z, s); s = fmaf(a.w, b.w, s);
        }
        sc[t] = s * scale;
    }
    __syncthreads();
    float m = -3.0e38f;
    for (int t = threadIdx.x; t < TSELF; t += blockDim.x) m = fmaxf(m, sc[t]);
    m = block_max(m, red);
    float ssum = 0.0f;
    for (int t = threadIdx.x; t < TSELF; t += blockDim.x) {
        float e = __expf(sc[t] - m); sc[t] = e; ssum += e;
    }
    ssum = block_sum(ssum, red);
    const float inv = 1.0f / ssum;
    __syncthreads();
    for (int d = threadIdx.x; d < DH; d += blockDim.x) {
        float acc = 0.0f;
        for (int t = 0; t < PASTLEN; ++t)
            acc = fmaf(sc[t], pastV[((size_t)h * PASTLEN + t) * DH + d], acc);
        acc = fmaf(sc[PASTLEN], vnew[h * DH + d], acc);
        o[h * DH + d] = acc * inv;
    }
}

// Kc = enc @ Wk, Vc = enc @ Wv   (grid.z selects K/V) — f16 WMMA, f32 accumulate.
// grid (ENCPAD/16, DMODEL/128, 2), block 256 (8 waves; wave w owns C tile cols w*16..w*16+15).
// LDS: A tile 16xKCHUNK, B tile stored TRANSPOSED [n][k] so each lane's fragment is
// two contiguous 16B chunks -> ds_load_b128 instead of scalar half loads.
__global__ void k_enc_kv_wmma(const float* __restrict__ enc,
                              const float* __restrict__ Wk, const float* __restrict__ Wv,
                              float* __restrict__ Kc, float* __restrict__ Vc) {
    const float* __restrict__ W = (blockIdx.z == 0) ? Wk : Wv;
    float* __restrict__ C = (blockIdx.z == 0) ? Kc : Vc;
    const int mbase = blockIdx.x * 16;
    const int nbase = blockIdx.y * 128;
    __shared__ __align__(16) _Float16 As[16][KCHUNK];      // [m][k]  4 KB
    __shared__ __align__(16) _Float16 BsT[128][KCHUNK];    // [n][k] 32 KB
    const int tid  = threadIdx.x;
    const int wave = tid >> 5;
    const int lane = tid & 31;
    const int mrow = lane & 15;
    const int hi   = lane >> 4;              // 0: lanes 0-15, 1: lanes 16-31
    v8f acc = {};
    for (int k0 = 0; k0 < DMODEL; k0 += KCHUNK) {
        // stage A: 16 x 128 f32 -> f16 (zero-pad rows >= ENCLEN), float4 loads
        for (int idx = tid * 4; idx < 16 * KCHUNK; idx += 256 * 4) {
            const int r = idx >> 7, c = idx & 127;
            const int row = mbase + r;
            float4 v = make_float4(0.f, 0.f, 0.f, 0.f);
            if (row < ENCLEN) v = *(const float4*)(enc + (size_t)row * DMODEL + k0 + c);
            As[r][c + 0] = (_Float16)v.x; As[r][c + 1] = (_Float16)v.y;
            As[r][c + 2] = (_Float16)v.z; As[r][c + 3] = (_Float16)v.w;
        }
        // stage B: 128(K) x 128(N) f32 -> f16 transposed into BsT[n][k], float4 loads
        for (int idx = tid * 4; idx < 128 * KCHUNK; idx += 256 * 4) {
            const int r = idx >> 7;          // k offset
            const int c = idx & 127;         // n offset
            float4 v = *(const float4*)(W + (size_t)(k0 + r) * DMODEL + nbase + c);
            BsT[c + 0][r] = (_Float16)v.x; BsT[c + 1][r] = (_Float16)v.y;
            BsT[c + 2][r] = (_Float16)v.z; BsT[c + 3][r] = (_Float16)v.w;
        }
        __syncthreads();
        const int bcol = wave * 16 + mrow;
        #pragma unroll
        for (int s = 0; s < KCHUNK / 32; ++s) {   // 4 WMMA per staged chunk
            // A fragment (16-bit 16x32 layout): lanes 0-15 K{0..7,16..23}, 16-31 K{8..15,24..31}
            const v8h a0 = *(const v8h*)&As[mrow][s * 32 + hi * 8];
            const v8h a1 = *(const v8h*)&As[mrow][s * 32 + hi * 8 + 16];
            // B fragment (32x16): lanes 0-15 K{0..7,16..23}, 16-31 K{8..15,24..31}, col = lane&15
            const v8h b0 = *(const v8h*)&BsT[bcol][s * 32 + hi * 8];
            const v8h b1 = *(const v8h*)&BsT[bcol][s * 32 + hi * 8 + 16];
            const v16h a = __builtin_shufflevector(a0, a1, 0,1,2,3,4,5,6,7,8,9,10,11,12,13,14,15);
            const v16h b = __builtin_shufflevector(b0, b1, 0,1,2,3,4,5,6,7,8,9,10,11,12,13,14,15);
            acc = __builtin_amdgcn_wmma_f32_16x16x32_f16(
                false, a, false, b, (short)0, acc, false, false);
        }
        __syncthreads();
    }
    // C fragment: VGPR r -> M = hi*8 + r, N = lane&15
    #pragma unroll
    for (int r = 0; r < 8; ++r) {
        const int row = mbase + hi * 8 + r;
        const int col = nbase + wave * 16 + mrow;
        C[(size_t)row * DMODEL + col] = acc[r];
    }
}

// cross-attention: one head per block over ENCLEN encoder positions
__global__ void k_cross_attn(const float* __restrict__ q, const float* __restrict__ Kc,
                             const float* __restrict__ Vc, float* __restrict__ o) {
    const int h = blockIdx.x;
    __shared__ __align__(16) float qs[DH];
    __shared__ float sc[ENCLEN];
    __shared__ float red[32];
    stage_to_lds_f32(qs, q + h * DH, DH);
    const float scale = 0.08838834764831843f;
    const float4* __restrict__ q4 = (const float4*)qs;
    for (int t = threadIdx.x; t < ENCLEN; t += blockDim.x) {
        const float4* __restrict__ k4 = (const float4*)(Kc + (size_t)t * DMODEL + h * DH);
        float s = 0.0f;
        #pragma unroll
        for (int d = 0; d < DH / 4; ++d) {
            float4 a = q4[d], b = k4[d];
            s = fmaf(a.x, b.x, s); s = fmaf(a.y, b.y, s);
            s = fmaf(a.z, b.z, s); s = fmaf(a.w, b.w, s);
        }
        sc[t] = s * scale;
    }
    __syncthreads();
    float m = -3.0e38f;
    for (int t = threadIdx.x; t < ENCLEN; t += blockDim.x) m = fmaxf(m, sc[t]);
    m = block_max(m, red);
    float ssum = 0.0f;
    for (int t = threadIdx.x; t < ENCLEN; t += blockDim.x) {
        float e = __expf(sc[t] - m); sc[t] = e; ssum += e;
    }
    ssum = block_sum(ssum, red);
    const float inv = 1.0f / ssum;
    __syncthreads();
    for (int d = threadIdx.x; d < DH; d += blockDim.x) {
        float acc = 0.0f;
        for (int t = 0; t < ENCLEN; ++t)
            acc = fmaf(sc[t], Vc[(size_t)t * DMODEL + h * DH + d], acc);
        o[h * DH + d] = acc * inv;
    }
}

// logits[v] = dot(xf, emb[v, :]) — one wave per vocab row, float4 streaming
__global__ void k_logits(const float* __restrict__ xf, const float* __restrict__ emb,
                         float* __restrict__ out) {
    __shared__ __align__(16) float xs[DMODEL];
    stage_to_lds_f32(xs, xf, DMODEL);
    const int wave = threadIdx.x >> 5, lane = threadIdx.x & 31;
    const int v = blockIdx.x * (blockDim.x >> 5) + wave;
    if (v >= VOCAB) return;
    const float4* __restrict__ e4 = (const float4*)(emb + (size_t)v * DMODEL);
    const float4* __restrict__ x4 = (const float4*)xs;
    float p = 0.0f;
    #pragma unroll
    for (int it = 0; it < DMODEL / 128; ++it) {   // 8 x float4 per lane
        const float4 ev = e4[it * 32 + lane];
        const float4 xv = x4[it * 32 + lane];
        p = fmaf(ev.x, xv.x, p); p = fmaf(ev.y, xv.y, p);
        p = fmaf(ev.z, xv.z, p); p = fmaf(ev.w, xv.w, p);
    }
    p = wave_sum(p);
    if (lane == 0) out[v] = p;
}

// ---------------- host-side orchestration ----------------
static inline void gemv(const float* h, const float* W, float* out, float* part,
                        int Din, int Nout, int mode, hipStream_t stream) {
    dim3 g(Nout / 256, GEMV_KS);
    k_gemv_part<<<g, 256, (Din / GEMV_KS) * sizeof(float), stream>>>(h, W, part, Din, Nout);
    k_gemv_fin<<<Nout / 256, 256, 0, stream>>>(part, out, Nout, mode);
}

extern "C" void kernel_launch(void* const* d_in, const int* in_sizes, int n_in,
                              void* d_out, int out_size, void* d_ws, size_t ws_size,
                              hipStream_t stream) {
    (void)in_sizes; (void)n_in; (void)out_size; (void)ws_size;
    const int*   ids   = (const int*)  d_in[0];
    const float* enc   = (const float*)d_in[1];
    const float* pastK = (const float*)d_in[2];
    const float* pastV = (const float*)d_in[3];
    const float* emb   = (const float*)d_in[4];
    const float* pos   = (const float*)d_in[5];
    const float* ln1_g = (const float*)d_in[6];
    const float* ln1_b = (const float*)d_in[7];
    const float* wq_s  = (const float*)d_in[8];
    const float* wk_s  = (const float*)d_in[9];
    const float* wv_s  = (const float*)d_in[10];
    const float* wo_s  = (const float*)d_in[11];
    const float* ln2_g = (const float*)d_in[12];
    const float* ln2_b = (const float*)d_in[13];
    const float* wq_c  = (const float*)d_in[14];
    const float* wk_c  = (const float*)d_in[15];
    const float* wv_c  = (const float*)d_in[16];
    const float* wo_c  = (const float*)d_in[17];
    const float* ln3_g = (const float*)d_in[18];
    const float* ln3_b = (const float*)d_in[19];
    const float* w1    = (const float*)d_in[20];
    const float* w2    = (const float*)d_in[21];
    const float* lnf_g = (const float*)d_in[22];
    const float* lnf_b = (const float*)d_in[23];
    const int*   plen  = (const int*)  d_in[24];

    float* ws  = (float*)d_ws;
    float* x    = ws;                  // 1024
    float* h    = ws + 1024;           // 1024
    float* q    = ws + 2048;           // 1024
    float* kk   = ws + 3072;           // 1024
    float* vv   = ws + 4096;           // 1024
    float* o    = ws + 5120;           // 1024
    float* ffn  = ws + 6144;           // 4096
    float* part = ws + 10240;          // GEMV_KS * 4096 = 32768
    float* Kc   = ws + 10240 + GEMV_KS * FFN;   // 448*1024
    float* Vc   = Kc + (size_t)ENCPAD * DMODEL; // 448*1024

    const size_t DD = (size_t)DMODEL * DMODEL;

    k_embed<<<4, 256, 0, stream>>>(ids, plen, emb, pos, x);

    for (int l = 0; l < LAYERS; ++l) {
        // ---- self attention ----
        k_layernorm<<<1, 256, 0, stream>>>(x, ln1_g + l * DMODEL, ln1_b + l * DMODEL, h, DMODEL);
        gemv(h, wq_s + l * DD, q,  part, DMODEL, DMODEL, 0, stream);
        gemv(h, wk_s + l * DD, kk, part, DMODEL, DMODEL, 0, stream);
        gemv(h, wv_s + l * DD, vv, part, DMODEL, DMODEL, 0, stream);
        k_self_attn<<<HEADS, 256, 0, stream>>>(q, kk, vv,
            pastK + (size_t)l * HEADS * PASTLEN * DH,
            pastV + (size_t)l * HEADS * PASTLEN * DH, o);
        gemv(o, wo_s + l * DD, x, part, DMODEL, DMODEL, 1, stream);

        // ---- cross attention ----
        k_layernorm<<<1, 256, 0, stream>>>(x, ln2_g + l * DMODEL, ln2_b + l * DMODEL, h, DMODEL);
        gemv(h, wq_c + l * DD, q, part, DMODEL, DMODEL, 0, stream);
        dim3 gkv(ENCPAD / 16, DMODEL / 128, 2);
        k_enc_kv_wmma<<<gkv, 256, 0, stream>>>(enc, wk_c + l * DD, wv_c + l * DD, Kc, Vc);
        k_cross_attn<<<HEADS, 256, 0, stream>>>(q, Kc, Vc, o);
        gemv(o, wo_c + l * DD, x, part, DMODEL, DMODEL, 1, stream);

        // ---- FFN ----
        k_layernorm<<<1, 256, 0, stream>>>(x, ln3_g + l * DMODEL, ln3_b + l * DMODEL, h, DMODEL);
        gemv(h,   w1 + (size_t)l * DMODEL * FFN, ffn, part, DMODEL, FFN, 2, stream);
        gemv(ffn, w2 + (size_t)l * FFN * DMODEL, x,   part, FFN, DMODEL, 1, stream);
    }

    k_layernorm<<<1, 256, 0, stream>>>(x, lnf_g, lnf_b, h, DMODEL);
    k_logits<<<VOCAB / 8, 256, 0, stream>>>(h, emb, (float*)d_out);
}
